// DGCNN_Transformer_79242146611366
// MI455X (gfx1250) — matrix operand order repaired and checked by reading, not compiled, not run on hardware
//
#include <hip/hip_runtime.h>
#include <stdint.h>

// ---------------------------------------------------------------------------
// DGCNN + grouped-attention edge convs for MI455X (gfx1250, wave32, WMMA).
// All GEMM-shaped math goes through v_wmma_f32_16x16x32_bf16 with 64x64
// macro-tiles (4 waves x 4 row-blocks, 8 WMMAs per wave per K-block).
// Workspace requirement: ~210 MB (laid out in kernel_launch).
// ---------------------------------------------------------------------------

#define NPTS  1024
#define BATCH 32
#define NCOLS 32768            // BATCH * NPTS
#define KNN   20
#define BNSF  0.9999950000374997f   // 1/sqrt(1+1e-5)
#define TP    72               // padded LDS tile row length (halves): 144B rows

typedef __attribute__((ext_vector_type(16))) __bf16 v16bf;
typedef __attribute__((ext_vector_type(8)))  __bf16 v8bf;
typedef __attribute__((ext_vector_type(8)))  float  v8f;

__device__ __forceinline__ unsigned short f2bf(float f) {
    unsigned u = __float_as_uint(f);
    unsigned r = u + 0x7FFFu + ((u >> 16) & 1u);   // round-to-nearest-even
    return (unsigned short)(r >> 16);
}
__device__ __forceinline__ float lrelu(float x) { return x >= 0.f ? x : 0.2f * x; }

// Stage a channel-pair (rows c, c+1 of the K-major source) x 8 consecutive
// columns, transposed into tile[j][c] with packed 32-bit LDS stores.
__device__ __forceinline__ void stage_tr_pair(const unsigned short* __restrict__ src0,
                                              const unsigned short* __restrict__ src1,
                                              unsigned short (*tile)[TP], int jb, int c) {
    uint4 va = *(const uint4*)src0;     // 8 bf16 of row c
    uint4 vb = *(const uint4*)src1;     // 8 bf16 of row c+1
    unsigned pa[4] = {va.x, va.y, va.z, va.w};
    unsigned pb[4] = {vb.x, vb.y, vb.z, vb.w};
    #pragma unroll
    for (int q = 0; q < 4; ++q) {
        unsigned lo0 = pa[q] & 0xffffu, hi0 = pa[q] >> 16;
        unsigned lo1 = pb[q] & 0xffffu, hi1 = pb[q] >> 16;
        *(unsigned*)&tile[jb + 2 * q][c]     = lo0 | (lo1 << 16);
        *(unsigned*)&tile[jb + 2 * q + 1][c] = hi0 | (hi1 << 16);
    }
}

// Stage 16 consecutive f32 weights of one A row, converted to bf16.
__device__ __forceinline__ void stage_w_row(const float* __restrict__ wsrc,
                                            unsigned short* __restrict__ dst) {
    #pragma unroll
    for (int q = 0; q < 4; ++q) {
        float4 v = *(const float4*)(wsrc + 4 * q);
        *(unsigned*)(dst + 4 * q)     = (unsigned)f2bf(v.x) | ((unsigned)f2bf(v.y) << 16);
        *(unsigned*)(dst + 4 * q + 2) = (unsigned)f2bf(v.z) | ((unsigned)f2bf(v.w) << 16);
    }
}

// Build A fragment (16x32 bf16) for K-substep s from one padded tile row.
__device__ __forceinline__ v16bf frag_a_row(const unsigned short* __restrict__ row,
                                            int h, int s) {
    v8bf alo = *(const v8bf*)(row + s * 32 + h * 16);
    v8bf ahi = *(const v8bf*)(row + s * 32 + h * 16 + 8);
    v16bf a;
    #pragma unroll
    for (int e = 0; e < 8; ++e) { a[e] = alo[e]; a[e + 8] = ahi[e]; }
    return a;
}

// ---------------------------------------------------------------------------
// x0 [32][3][1024] f32  ->  xb [3][NCOLS] bf16 (channel-major), xx [NCOLS]
// ---------------------------------------------------------------------------
__global__ void convert_x0_kernel(const float* __restrict__ x0,
                                  unsigned short* __restrict__ xb,
                                  float* __restrict__ xx) {
    int col = blockIdx.x * blockDim.x + threadIdx.x;
    if (col >= NCOLS) return;
    int b = col >> 10, n = col & 1023;
    float s = 0.f;
    for (int c = 0; c < 3; ++c) {
        float v = x0[(b * 3 + c) * NPTS + n];
        xb[c * NCOLS + col] = f2bf(v);
        s += v * v;
    }
    xx[col] = s;
}

// ---------------------------------------------------------------------------
// wk [O][2C] -> wsum = wk[:, :C] + wk[:, C:],  wd = wk[:, :C]
// ---------------------------------------------------------------------------
__global__ void prep_w_kernel(const float* __restrict__ w,
                              float* __restrict__ wsum, float* __restrict__ wd,
                              int O, int C) {
    int i = blockIdx.x * blockDim.x + threadIdx.x;
    if (i >= O * C) return;
    int o = i / C, c = i - o * C;
    float a  = w[o * 2 * C + c];
    float bb = w[o * 2 * C + C + c];
    wd[i]   = a;
    wsum[i] = a + bb;
}

// ---------------------------------------------------------------------------
// Pairwise distances for one batch: pd[n][m] = 2*inner - xx[n] - xx[m].
// 64x64 macro-tile: rows n0..n0+63 (4 row-blocks), wave w -> cols m0+16w..+15.
// K blocked by 64: one staging phase feeds 8 WMMAs per wave.
// ---------------------------------------------------------------------------
__global__ void __launch_bounds__(128)
dist_wmma_kernel(const unsigned short* __restrict__ xb,
                 const float* __restrict__ xx,
                 float* __restrict__ pd, int C, int colbase) {
    __shared__ unsigned short At[64][TP];   // [rows n][K=c]   (transposed vs memory)
    __shared__ unsigned short Bt[64][TP];   // [cols m][K=c]   (transposed vs memory)
    int t = threadIdx.x, wave = t >> 5, lane = t & 31;
    int n0 = blockIdx.y * 64;
    int m0 = blockIdx.x * 64;
    int abase = colbase + n0;
    int bbase = colbase + m0;
    v8f acc[4] = {};
    int ksteps = (C + 63) >> 6;
    for (int kk = 0; kk < ksteps; ++kk) {
        int c0 = kk << 6;
        if (c0 + 64 <= C) {                       // uniform fast path: no divergence
            #pragma unroll
            for (int it = 0; it < 2; ++it) {      // A: 64r x 64c = 256 pair-tasks
                int idx = t + it * 128;
                int c = (idx & 31) * 2, rb = (idx >> 5) * 8;
                stage_tr_pair(xb + (size_t)(c0 + c) * NCOLS + abase + rb,
                              xb + (size_t)(c0 + c + 1) * NCOLS + abase + rb,
                              At, rb, c);
            }
            #pragma unroll
            for (int it = 0; it < 2; ++it) {      // B: 64j x 64c = 256 pair-tasks
                int idx = t + it * 128;
                int c = (idx & 31) * 2, jb = (idx >> 5) * 8;
                stage_tr_pair(xb + (size_t)(c0 + c) * NCOLS + bbase + jb,
                              xb + (size_t)(c0 + c + 1) * NCOLS + bbase + jb,
                              Bt, jb, c);
            }
            if (c0 + 64 < C) {                    // prefetch next K-block B panel
                int cc = c0 + 64 + (t >> 1);
                if (cc < C)
                    __builtin_prefetch(xb + (size_t)cc * NCOLS + bbase + (t & 1) * 32, 0, 0);
            }
        } else {                                  // rare guarded path (C==3 only)
            for (int i = t; i < 4096; i += 128) {
                int c = i & 63, r = i >> 6, cc = c0 + c;
                At[r][c] = (cc < C) ? xb[(size_t)cc * NCOLS + abase + r] : (unsigned short)0;
            }
            for (int i = t; i < 4096; i += 128) {
                int c = i & 63, j = i >> 6, cc = c0 + c;
                Bt[j][c] = (cc < C) ? xb[(size_t)cc * NCOLS + bbase + j] : (unsigned short)0;
            }
        }
        __syncthreads();
        int r = lane & 15, h = lane >> 4;
        #pragma unroll
        for (int s = 0; s < 2; ++s) {
            v16bf b = *(const v16bf*)&Bt[wave * 16 + r][s * 32 + h * 16];
            #pragma unroll
            for (int q = 0; q < 4; ++q) {
                v16bf a = frag_a_row(At[q * 16 + r], h, s);
                acc[q] = __builtin_amdgcn_wmma_f32_16x16x32_bf16(false, a, false, b,
                                                                 (short)0, acc[q],
                                                                 false, false);
            }
        }
        __syncthreads();
    }
    int r = lane & 15, h = lane >> 4;
    int m = m0 + wave * 16 + r;
    float xxm = xx[colbase + m];
    #pragma unroll
    for (int q = 0; q < 4; ++q) {
        #pragma unroll
        for (int e = 0; e < 8; ++e) {
            int n = n0 + q * 16 + e + h * 8;
            pd[n * NPTS + m] = 2.f * acc[q][e] - xx[colbase + n] - xxm;
        }
    }
}

// ---------------------------------------------------------------------------
// Top-K=20 (largest pd) per row; one thread per row; register-resident lists.
// ---------------------------------------------------------------------------
__global__ void topk_kernel(const float* __restrict__ pd, int* __restrict__ idxout, int b) {
    int n = blockIdx.x * blockDim.x + threadIdx.x;
    if (n >= NPTS) return;
    const float* row = pd + n * NPTS;
    float vals[KNN]; int idxs[KNN];
    #pragma unroll
    for (int j = 0; j < KNN; ++j) { vals[j] = row[j]; idxs[j] = j; }
    float minv = vals[0]; int minp = 0;
    #pragma unroll
    for (int j = 1; j < KNN; ++j) if (vals[j] < minv) { minv = vals[j]; minp = j; }
    for (int m = KNN; m < NPTS; ++m) {
        float v = row[m];
        if (v > minv) {
            #pragma unroll
            for (int j = 0; j < KNN; ++j) if (j == minp) { vals[j] = v; idxs[j] = m; }
            minv = vals[0]; minp = 0;
            #pragma unroll
            for (int j = 1; j < KNN; ++j) if (vals[j] < minv) { minv = vals[j]; minp = j; }
        }
    }
    int* orow = idxout + (b * NPTS + n) * KNN;
    #pragma unroll
    for (int j = 0; j < KNN; ++j) orow[j] = idxs[j];
}

// ---------------------------------------------------------------------------
// out[M][NCOLS] = W[M][K](f32) * Xb[K][NCOLS](bf16), 64x64 macro-tile.
// mode 0: store f32.  mode 1 (conv5): leaky(acc*BNS), lane-reduce max over
// 16 cols, atomicMax into order-preserving uint pool[b][m].
// ---------------------------------------------------------------------------
__global__ void __launch_bounds__(128)
gemm_wmma_kernel(const float* __restrict__ W, const unsigned short* __restrict__ Xb,
                 float* __restrict__ out, unsigned int* __restrict__ pool,
                 int M, int Kdim, int mode) {
    __shared__ unsigned short At[64][TP];
    __shared__ unsigned short Bt[64][TP];
    int t = threadIdx.x, wave = t >> 5, lane = t & 31;
    int col0 = blockIdx.x * 64;
    int m0 = blockIdx.y * 64;
    v8f acc[4] = {};
    int ksteps = (Kdim + 63) >> 6;
    for (int kk = 0; kk < ksteps; ++kk) {
        int c0 = kk << 6;
        if (c0 + 64 <= Kdim) {                    // uniform fast path
            #pragma unroll
            for (int it = 0; it < 2; ++it) {      // A: 64 rows x 4 16-col chunks
                int idx = t + it * 128;
                int r = idx & 63, cb = (idx >> 6) * 16;
                stage_w_row(W + (size_t)(m0 + r) * Kdim + c0 + cb, &At[r][cb]);
            }
            #pragma unroll
            for (int it = 0; it < 2; ++it) {      // B: 64j x 64c = 256 pair-tasks
                int idx = t + it * 128;
                int c = (idx & 31) * 2, jb = (idx >> 5) * 8;
                stage_tr_pair(Xb + (size_t)(c0 + c) * NCOLS + col0 + jb,
                              Xb + (size_t)(c0 + c + 1) * NCOLS + col0 + jb,
                              Bt, jb, c);
            }
            if (c0 + 64 < Kdim) {
                int cc = c0 + 64 + (t >> 1);
                if (cc < Kdim)
                    __builtin_prefetch(Xb + (size_t)cc * NCOLS + col0 + (t & 1) * 32, 0, 0);
            }
        } else {                                  // rare guarded path (K==3 only)
            for (int i = t; i < 4096; i += 128) {
                int c = i & 63, r = i >> 6, cc = c0 + c;
                At[r][c] = (cc < Kdim) ? f2bf(W[(size_t)(m0 + r) * Kdim + cc])
                                       : (unsigned short)0;
            }
            for (int i = t; i < 4096; i += 128) {
                int c = i & 63, j = i >> 6, cc = c0 + c;
                Bt[j][c] = (cc < Kdim) ? Xb[(size_t)cc * NCOLS + col0 + j]
                                       : (unsigned short)0;
            }
        }
        __syncthreads();
        int r = lane & 15, h = lane >> 4;
        #pragma unroll
        for (int s = 0; s < 2; ++s) {
            v16bf b = *(const v16bf*)&Bt[wave * 16 + r][s * 32 + h * 16];
            #pragma unroll
            for (int q = 0; q < 4; ++q) {
                v16bf a = frag_a_row(At[q * 16 + r], h, s);
                acc[q] = __builtin_amdgcn_wmma_f32_16x16x32_bf16(false, a, false, b,
                                                                 (short)0, acc[q],
                                                                 false, false);
            }
        }
        __syncthreads();
    }
    int r = lane & 15, h = lane >> 4;
    int col = col0 + wave * 16 + r;
    if (mode == 0) {
        #pragma unroll
        for (int q = 0; q < 4; ++q) {
            #pragma unroll
            for (int e = 0; e < 8; ++e)
                out[(size_t)(m0 + q * 16 + e + h * 8) * NCOLS + col] = acc[q][e];
        }
    } else {
        int b = col >> 10;   // 64-col tile never straddles a batch (1024%64==0)
        #pragma unroll
        for (int q = 0; q < 4; ++q) {
            #pragma unroll
            for (int e = 0; e < 8; ++e) {
                float v = lrelu(acc[q][e] * BNSF);
                #pragma unroll
                for (int s = 1; s < 16; s <<= 1) v = fmaxf(v, __shfl_xor(v, s, 32));
                if (r == 0) {
                    unsigned ub = __float_as_uint(v);
                    unsigned enc = (ub & 0x80000000u) ? ~ub : (ub | 0x80000000u);
                    atomicMax(&pool[b * NPTS + (m0 + q * 16 + e + h * 8)], enc);
                }
            }
        }
    }
}

// ---------------------------------------------------------------------------
// Grouped attention per column (b,n). k/v reconstructed via
// k(o,j) = PkS[o, idx_j] - QkD[o, n]. Writes next-layer bf16 acts + sq-norms.
// ---------------------------------------------------------------------------
__global__ void __launch_bounds__(64)
attn_kernel(const float* __restrict__ Pq,  const float* __restrict__ PkS,
            const float* __restrict__ QkD, const float* __restrict__ PvS,
            const float* __restrict__ QvD, const int* __restrict__ idxbuf,
            unsigned short* __restrict__ xbOut, float* __restrict__ xxOut, int O) {
    __shared__ int   cols[KNN];
    __shared__ float qL[256], QkL[256], QvL[256];
    __shared__ float en[8][KNN];
    __shared__ float red[64];
    int t = threadIdx.x;
    int col = blockIdx.x;
    int b = col >> 10;
    int gd = O >> 3;
    if (t < KNN) cols[t] = (b << 10) + idxbuf[col * KNN + t];
    for (int o = t; o < O; o += 64) {
        qL[o]  = Pq[(size_t)o * NCOLS + col];
        QkL[o] = QkD[(size_t)o * NCOLS + col];
        QvL[o] = QvD[(size_t)o * NCOLS + col];
    }
    __syncthreads();
    if (t < KNN) {                            // thread j: energies over 8 groups
        int cj = cols[t];
        float rs = rsqrtf((float)gd);
        for (int g = 0; g < 8; ++g) {
            float e = 0.f;
            int base = g * gd;
            for (int d = 0; d < gd; ++d) {
                int o = base + d;
                e += qL[o] * (PkS[(size_t)o * NCOLS + cj] - QkL[o]);
            }
            en[g][t] = e * rs;
        }
    }
    __syncthreads();
    if (t < 8) {                              // softmax per group
        float mx = en[t][0];
        for (int j = 1; j < KNN; ++j) mx = fmaxf(mx, en[t][j]);
        float s = 0.f;
        for (int j = 0; j < KNN; ++j) { float w = __expf(en[t][j] - mx); en[t][j] = w; s += w; }
        float inv = 1.f / s;
        for (int j = 0; j < KNN; ++j) en[t][j] *= inv;
    }
    __syncthreads();
    float ss = 0.f;
    for (int o = t; o < O; o += 64) {
        int g = o / gd;
        float acc = 0.f;
        for (int j = 0; j < KNN; ++j)
            acc += en[g][j] * (PvS[(size_t)o * NCOLS + cols[j]] - QvL[o]);
        float v = lrelu(acc * BNSF);
        xbOut[(size_t)o * NCOLS + col] = f2bf(v);
        ss += v * v;
    }
    red[t] = ss;
    __syncthreads();
    if (t == 0) {
        float s = 0.f;
        for (int i = 0; i < 64; ++i) s += red[i];
        xxOut[col] = s;
    }
}

__global__ void pool_init_kernel(unsigned int* __restrict__ pool) {
    int i = blockIdx.x * blockDim.x + threadIdx.x;
    if (i < BATCH * NPTS) pool[i] = 0u;       // below any encoded finite float
}

// ---------------------------------------------------------------------------
// Head: decode pooled max -> lin1+bn+leaky -> lin2+bias+bn+leaky -> lin3+bias
// ---------------------------------------------------------------------------
__global__ void __launch_bounds__(256)
head_kernel(const unsigned int* __restrict__ pool,
            const float* __restrict__ w1, const float* __restrict__ w2,
            const float* __restrict__ b2, const float* __restrict__ w3,
            const float* __restrict__ b3, float* __restrict__ out) {
    __shared__ float h[1024], h1[512], h2[256];
    int b = blockIdx.x, t = threadIdx.x;
    for (int i = t; i < 1024; i += 256) {
        unsigned enc = pool[b * NPTS + i];
        unsigned ub = (enc & 0x80000000u) ? (enc & 0x7FFFFFFFu) : ~enc;
        h[i] = __uint_as_float(ub);
    }
    __syncthreads();
    for (int o = t; o < 512; o += 256) {
        float acc = 0.f;
        const float* wr = w1 + o * 1024;
        for (int i = 0; i < 1024; ++i) acc += wr[i] * h[i];
        h1[o] = lrelu(acc * BNSF);
    }
    __syncthreads();
    {
        float acc = b2[t];
        const float* wr = w2 + t * 512;
        for (int i = 0; i < 512; ++i) acc += wr[i] * h1[i];
        h2[t] = lrelu(acc * BNSF);
    }
    __syncthreads();
    if (t < 40) {
        float acc = b3[t];
        const float* wr = w3 + t * 256;
        for (int i = 0; i < 256; ++i) acc += wr[i] * h2[i];
        out[b * 40 + t] = acc;
    }
}

// ---------------------------------------------------------------------------
extern "C" void kernel_launch(void* const* d_in, const int* in_sizes, int n_in,
                              void* d_out, int out_size, void* d_ws, size_t ws_size,
                              hipStream_t stream) {
    (void)in_sizes; (void)n_in; (void)out_size; (void)ws_size;

    const float* x0 = (const float*)d_in[0];
    const float* wq[4] = {(const float*)d_in[1], (const float*)d_in[4],
                          (const float*)d_in[7], (const float*)d_in[10]};
    const float* wk[4] = {(const float*)d_in[2], (const float*)d_in[5],
                          (const float*)d_in[8], (const float*)d_in[11]};
    const float* wv[4] = {(const float*)d_in[3], (const float*)d_in[6],
                          (const float*)d_in[9], (const float*)d_in[12]};
    const float* conv5w = (const float*)d_in[13];
    const float* lin1w  = (const float*)d_in[14];
    const float* lin2w  = (const float*)d_in[15];
    const float* lin2b  = (const float*)d_in[16];
    const float* lin3w  = (const float*)d_in[17];
    const float* lin3b  = (const float*)d_in[18];

    const int Cin[4] = {3, 64, 64, 128};
    const int Oc[4]  = {64, 64, 128, 256};

    // ---- workspace layout (~210 MB) ----
    size_t off = 0;
    auto alloc = [&](size_t bytes) -> char* {
        char* p = (char*)d_ws + off;
        off = (off + bytes + 255) & ~(size_t)255;
        return p;
    };
    unsigned short* XB[2];
    XB[0] = (unsigned short*)alloc(256UL * NCOLS * 2);
    XB[1] = (unsigned short*)alloc(256UL * NCOLS * 2);
    float* XX   = (float*)alloc((size_t)NCOLS * 4);
    float* Pq   = (float*)alloc(256UL * NCOLS * 4);
    float* PkS  = (float*)alloc(256UL * NCOLS * 4);
    float* QkD  = (float*)alloc(256UL * NCOLS * 4);
    float* PvS  = (float*)alloc(256UL * NCOLS * 4);
    float* QvD  = (float*)alloc(256UL * NCOLS * 4);
    float* PD   = (float*)alloc((size_t)NPTS * NPTS * 4);
    int*   IDX  = (int*)alloc((size_t)BATCH * NPTS * KNN * 4);
    unsigned int* POOL = (unsigned int*)alloc((size_t)BATCH * NPTS * 4);
    float* WkS = (float*)alloc(256UL * 256 * 4);
    float* WkD = (float*)alloc(256UL * 256 * 4);
    float* WvS = (float*)alloc(256UL * 256 * 4);
    float* WvD = (float*)alloc(256UL * 256 * 4);

    convert_x0_kernel<<<NCOLS / 256, 256, 0, stream>>>(x0, XB[0], XX);

    int cur = 0;
    for (int l = 0; l < 4; ++l) {
        int C = Cin[l], O = Oc[l];
        prep_w_kernel<<<(O * C + 255) / 256, 256, 0, stream>>>(wk[l], WkS, WkD, O, C);
        prep_w_kernel<<<(O * C + 255) / 256, 256, 0, stream>>>(wv[l], WvS, WvD, O, C);

        for (int b = 0; b < BATCH; ++b) {
            dist_wmma_kernel<<<dim3(NPTS / 64, NPTS / 64), 128, 0, stream>>>(
                XB[cur], XX, PD, C, b * NPTS);
            topk_kernel<<<NPTS / 256, 256, 0, stream>>>(PD, IDX, b);
        }

        dim3 ggrid(NCOLS / 64, O / 64);
        gemm_wmma_kernel<<<ggrid, 128, 0, stream>>>(wq[l], XB[cur], Pq,  nullptr, O, C, 0);
        gemm_wmma_kernel<<<ggrid, 128, 0, stream>>>(WkS,   XB[cur], PkS, nullptr, O, C, 0);
        gemm_wmma_kernel<<<ggrid, 128, 0, stream>>>(WkD,   XB[cur], QkD, nullptr, O, C, 0);
        gemm_wmma_kernel<<<ggrid, 128, 0, stream>>>(WvS,   XB[cur], PvS, nullptr, O, C, 0);
        gemm_wmma_kernel<<<ggrid, 128, 0, stream>>>(WvD,   XB[cur], QvD, nullptr, O, C, 0);

        attn_kernel<<<NCOLS, 64, 0, stream>>>(Pq, PkS, QkD, PvS, QvD, IDX,
                                              XB[1 - cur], XX, O);
        cur = 1 - cur;
    }

    pool_init_kernel<<<(BATCH * NPTS + 255) / 256, 256, 0, stream>>>(POOL);
    gemm_wmma_kernel<<<dim3(NCOLS / 64, 1024 / 64), 128, 0, stream>>>(
        conv5w, XB[cur], nullptr, POOL, 1024, 256, 1);
    head_kernel<<<BATCH, 256, 0, stream>>>(POOL, lin1w, lin2w, lin2b, lin3w, lin3b,
                                           (float*)d_out);
}
